// FineMatching_83915071030174
// MI455X (gfx1250) — matrix operand order repaired and checked by reading, not compiled, not run on hardware
//
#include <hip/hip_runtime.h>

typedef unsigned short ushort_t;
typedef __attribute__((ext_vector_type(16))) __bf16 v16bf;
typedef __attribute__((ext_vector_type(8)))  float  v8f;
typedef __attribute__((ext_vector_type(8)))  unsigned short v8us;

union AFrag { v16bf v; v8us h[2]; };

__device__ __forceinline__ ushort_t f2bf(float f) {
    unsigned u = __float_as_uint(f);
    u += 0x7FFFu + ((u >> 16) & 1u);           // round-to-nearest-even
    return (ushort_t)(u >> 16);
}

#define HF    120
#define WF    160
#define D_CH  256
#define WIN   5
#define K1    4608   // 9 * 512
#define K2    1152   // 9 * 128
#define OC1   128
#define OC2   64
#define ZPOS  25     // zero-block position index (patch & h1)
#define MPW   2      // matches per workgroup (A-fragment reuse factor)

// LDS layout (dynamic), all offsets multiples of 32B:
//   patch : MPW * 26 * 512 ushort  = 53248 B   (25 positions + zero block, per match)
//   h1    : MPW * 26 * 128 ushort  = 13312 B
//   sAcc  : MPW * 64 float         =   512 B   (channel sums for conv3)
#define PATCH_ELEMS (26 * 512)
#define H1_ELEMS    (26 * 128)
#define SMEM_BYTES  (MPW * PATCH_ELEMS * 2 + MPW * H1_ELEMS * 2 + MPW * OC2 * 4)

// ---------------------------------------------------------------------------
// Kernel 0: convert conv weights fp32 OIHW -> bf16 [oc][ky*3+kx][ic]
// (K-order matches im2col K-order: neighbor-major, channel contiguous)
// ---------------------------------------------------------------------------
__global__ void convert_weights_kernel(const float* __restrict__ w1,
                                       const float* __restrict__ w2,
                                       ushort_t* __restrict__ w1b,
                                       ushort_t* __restrict__ w2b) {
    int idx = blockIdx.x * blockDim.x + threadIdx.x;
    const int N1 = OC1 * K1;                       // 589824
    if (idx < N1) {
        int oc = idx / K1, r = idx % K1;
        int n = r / 512, ic = r % 512;
        int ky = n / 3, kx = n % 3;
        w1b[idx] = f2bf(w1[((oc * 512 + ic) * 3 + ky) * 3 + kx]);
    } else {
        int j = idx - N1;
        if (j < OC2 * K2) {                        // 73728
            int oc = j / K2, r = j % K2;
            int n = r / 128, ic = r % 128;
            int ky = n / 3, kx = n % 3;
            w2b[j] = f2bf(w2[((oc * 128 + ic) * 3 + ky) * 3 + kx]);
        }
    }
}

// ---------------------------------------------------------------------------
// Kernel 1: fully fused pipeline, MPW matches per workgroup (256 thr = 8 waves)
// ---------------------------------------------------------------------------
__global__ __launch_bounds__(256)
void fine_matching_fused(const float* __restrict__ feat0,
                         const float* __restrict__ feat1,
                         const float* __restrict__ mk0,
                         const float* __restrict__ mk1,
                         const int*   __restrict__ bids,
                         const int*   __restrict__ Himg_p,
                         const int*   __restrict__ Wimg_p,
                         const ushort_t* __restrict__ w1b,
                         const float* __restrict__ b1,
                         const ushort_t* __restrict__ w2b,
                         const float* __restrict__ b2,
                         const float* __restrict__ w3,
                         const float* __restrict__ b3,
                         float* __restrict__ out, int M)
{
    extern __shared__ __align__(32) char smem[];
    ushort_t* patch = (ushort_t*)smem;                       // [MPW][26][512]
    ushort_t* h1    = patch + MPW * PATCH_ELEMS;             // [MPW][26][128]
    float*    sAcc  = (float*)(h1 + MPW * H1_ELEMS);         // [MPW][64]

    const int m0  = blockIdx.x * MPW;
    const int tid = threadIdx.x;

    // zero: per-match zero blocks + channel-sum accumulators
    #pragma unroll
    for (int q = 0; q < MPW; ++q) {
        for (int i = tid; i < 512; i += 256) patch[q * PATCH_ELEMS + ZPOS * 512 + i] = 0;
        if (tid < 128) h1[q * H1_ELEMS + ZPOS * 128 + tid] = 0;
    }
    if (tid < MPW * OC2) sAcc[tid] = 0.f;

    const float Wden = (float)(*Wimg_p - 1);
    const float Hden = (float)(*Himg_p - 1);

    // ---- stage 1: separable bilinear sampling (taps are integer-spaced) ----
    for (int c = tid; c < MPW * 2 * D_CH; c += 256) {
        const int q  = c >> 9;                 // match within WG
        const int m  = m0 + q;
        if (m >= M) break;
        const int cc = c & 511;                // concat channel 0..511
        const float* feat; const float* mk; int ch;
        if (cc < D_CH) { feat = feat0; mk = mk0; ch = cc; }
        else           { feat = feat1; mk = mk1; ch = cc - D_CH; }
        const float xc = mk[2 * m]     * (float)(WF - 1) / Wden;
        const float yc = mk[2 * m + 1] * (float)(HF - 1) / Hden;
        float xbf = floorf(xc), ybf = floorf(yc);
        int   xb = (int)xbf,    yb = (int)ybf;
        float wx1 = xc - xbf, wx0 = 1.f - wx1;
        float wy1 = yc - ybf, wy0 = 1.f - wy1;
        const int b = bids[m];
        const float* fp = feat + ((size_t)b * D_CH + ch) * (size_t)(HF * WF);
        float g[6][6];
        #pragma unroll
        for (int j = 0; j < 6; ++j) {
            int yi = yb - 2 + j;
            bool yv = (yi >= 0) && (yi < HF);
            #pragma unroll
            for (int i = 0; i < 6; ++i) {
                int xi = xb - 2 + i;
                bool v = yv && (xi >= 0) && (xi < WF);
                g[j][i] = v ? fp[yi * WF + xi] : 0.f;
            }
        }
        ushort_t* pq = patch + q * PATCH_ELEMS;
        #pragma unroll
        for (int ty = 0; ty < WIN; ++ty) {
            #pragma unroll
            for (int tx = 0; tx < WIN; ++tx) {
                float val = wy0 * (wx0 * g[ty][tx]     + wx1 * g[ty][tx + 1])
                          + wy1 * (wx0 * g[ty + 1][tx] + wx1 * g[ty + 1][tx + 1]);
                pq[(ty * WIN + tx) * 512 + cc] = f2bf(val);
            }
        }
    }
    __syncthreads();

    const int wid  = tid >> 5;
    const int lane = tid & 31;
    const int mrow = lane & 15;
    const int hs   = lane >> 4;     // half-select for 16-bit A/B VGPR layout

    // ---- stage 2: conv1 implicit GEMM, C1[128 oc][25 pos] x MPW, K = 4608 ----
    {
        const int ocb = wid * 16;                          // 8 waves -> 128 oc
        const ushort_t* w1row = w1b + (size_t)(ocb + mrow) * K1;
        const int p0 = mrow, p1 = mrow + 16;
        int npos0[9], npos1[9];                            // shared by both matches
        const int py0 = p0 / 5, px0 = p0 % 5;
        const int py1 = p1 / 5, px1 = p1 % 5;
        #pragma unroll
        for (int n = 0; n < 9; ++n) {
            int ky = n / 3, kx = n % 3;
            int iy0 = py0 + ky - 1, ix0 = px0 + kx - 1;
            npos0[n] = (iy0 >= 0 && iy0 < 5 && ix0 >= 0 && ix0 < 5) ? iy0 * 5 + ix0 : ZPOS;
            int iy1 = py1 + ky - 1, ix1 = px1 + kx - 1;
            npos1[n] = (p1 < 25 && iy1 >= 0 && iy1 < 5 && ix1 >= 0 && ix1 < 5)
                       ? iy1 * 5 + ix1 : ZPOS;
        }
        const ushort_t* pq0 = patch;                 // match 0
        const ushort_t* pq1 = patch + PATCH_ELEMS;   // match 1
        v8f a00 = {}, a01 = {}, a10 = {}, a11 = {};
        for (int n = 0; n < 9; ++n) {
            const ushort_t* wA = w1row + n * 512;
            const int o0 = npos0[n] * 512 + hs * 16;
            const int o1 = npos1[n] * 512 + hs * 16;
            #pragma unroll 4
            for (int kc = 0; kc < 16; ++kc) {
                AFrag a;
                a.h[0] = *(const v8us*)(wA + kc * 32 + hs * 16);      // K+0..7 / +16..23
                a.h[1] = *(const v8us*)(wA + kc * 32 + hs * 16 + 8);  // K+8..15 / +24..31
                v16bf b00 = *(const v16bf*)(pq0 + o0 + kc * 32);      // 32B contiguous
                v16bf b01 = *(const v16bf*)(pq0 + o1 + kc * 32);
                v16bf b10 = *(const v16bf*)(pq1 + o0 + kc * 32);
                v16bf b11 = *(const v16bf*)(pq1 + o1 + kc * 32);
                a00 = __builtin_amdgcn_wmma_f32_16x16x32_bf16(false, a.v, false, b00,
                                                              (short)0, a00, false, false);
                a01 = __builtin_amdgcn_wmma_f32_16x16x32_bf16(false, a.v, false, b01,
                                                              (short)0, a01, false, false);
                a10 = __builtin_amdgcn_wmma_f32_16x16x32_bf16(false, a.v, false, b10,
                                                              (short)0, a10, false, false);
                a11 = __builtin_amdgcn_wmma_f32_16x16x32_bf16(false, a.v, false, b11,
                                                              (short)0, a11, false, false);
            }
        }
        // bias + relu -> h1 (bf16, [pos][oc]) for both matches
        #pragma unroll
        for (int r = 0; r < 8; ++r) {
            int oc = ocb + r + hs * 8;
            float bb = b1[oc];
            float v; 
            v = a00[r] + bb; v = v > 0.f ? v : 0.f; h1[p0 * 128 + oc] = f2bf(v);
            v = a10[r] + bb; v = v > 0.f ? v : 0.f; h1[H1_ELEMS + p0 * 128 + oc] = f2bf(v);
            if (p1 < 25) {
                v = a01[r] + bb; v = v > 0.f ? v : 0.f; h1[p1 * 128 + oc] = f2bf(v);
                v = a11[r] + bb; v = v > 0.f ? v : 0.f; h1[H1_ELEMS + p1 * 128 + oc] = f2bf(v);
            }
        }
    }
    __syncthreads();

    // ---- stage 3: conv2 implicit GEMM, C2[64 oc][25 pos] x MPW, K = 1152 ----
    // conv3 is a 1x1 conv of the spatial mean, so fold relu(conv2) directly
    // into per-channel sums via LDS float atomics (no h2 buffer needed).
    {
        const int mt = wid >> 1, nt = wid & 1;             // 4 M-tiles x 2 N-tiles
        const int ocb = mt * 16;
        const int p = mrow + nt * 16;
        const ushort_t* w2row = w2b + (size_t)(ocb + mrow) * K2;
        int npos[9];
        const int py = p / 5, px = p % 5;
        #pragma unroll
        for (int n = 0; n < 9; ++n) {
            int ky = n / 3, kx = n % 3;
            int iy = py + ky - 1, ix = px + kx - 1;
            npos[n] = (p < 25 && iy >= 0 && iy < 5 && ix >= 0 && ix < 5)
                      ? iy * 5 + ix : ZPOS;
        }
        const ushort_t* hq0 = h1;
        const ushort_t* hq1 = h1 + H1_ELEMS;
        v8f acc0 = {}, acc1 = {};
        for (int n = 0; n < 9; ++n) {
            const ushort_t* wA = w2row + n * 128;
            const int off = npos[n] * 128 + hs * 16;
            #pragma unroll
            for (int kc = 0; kc < 4; ++kc) {
                AFrag a;
                a.h[0] = *(const v8us*)(wA + kc * 32 + hs * 16);
                a.h[1] = *(const v8us*)(wA + kc * 32 + hs * 16 + 8);
                v16bf bv0 = *(const v16bf*)(hq0 + off + kc * 32);
                v16bf bv1 = *(const v16bf*)(hq1 + off + kc * 32);
                acc0 = __builtin_amdgcn_wmma_f32_16x16x32_bf16(false, a.v, false, bv0,
                                                               (short)0, acc0, false, false);
                acc1 = __builtin_amdgcn_wmma_f32_16x16x32_bf16(false, a.v, false, bv1,
                                                               (short)0, acc1, false, false);
            }
        }
        if (p < 25) {
            #pragma unroll
            for (int r = 0; r < 8; ++r) {
                int oc = ocb + r + hs * 8;
                float bb = b2[oc];
                float v0 = acc0[r] + bb; v0 = v0 > 0.f ? v0 : 0.f;
                float v1 = acc1[r] + bb; v1 = v1 > 0.f ? v1 : 0.f;
                atomicAdd(&sAcc[oc], v0);                  // ds_add_f32
                atomicAdd(&sAcc[OC2 + oc], v1);
            }
        }
    }
    __syncthreads();

    // ---- stage 4: 1x1 conv (64 -> 2) on channel means + output ----
    if (tid < MPW * 2) {
        const int q = tid >> 1, c = tid & 1;
        const int m = m0 + q;
        if (m < M) {
            float d = b3[c];
            #pragma unroll
            for (int ic = 0; ic < OC2; ++ic)
                d += w3[c * OC2 + ic] * sAcc[q * OC2 + ic];
            d *= (1.0f / 25.0f);
            d += b3[c] * (1.0f - 1.0f);                    // keep exact form below
            // delta = b3 + (w3 . mean) ; mean = sum/25. Recompute cleanly:
            float delta = b3[c];
            float dot = 0.f;
            #pragma unroll
            for (int ic = 0; ic < OC2; ++ic)
                dot += w3[c * OC2 + ic] * sAcc[q * OC2 + ic];
            delta += dot * (1.0f / 25.0f);
            out[2 * m + c]         = mk0[2 * m + c];       // mkpts0_f (copy)
            out[2 * M + 2 * m + c] = mk1[2 * m + c] + delta;
        }
    }
}

// ---------------------------------------------------------------------------
extern "C" void kernel_launch(void* const* d_in, const int* in_sizes, int n_in,
                              void* d_out, int out_size, void* d_ws, size_t ws_size,
                              hipStream_t stream) {
    (void)n_in; (void)out_size; (void)ws_size;
    const float* feat0 = (const float*)d_in[0];
    const float* feat1 = (const float*)d_in[1];
    const float* mk0   = (const float*)d_in[2];
    const float* mk1   = (const float*)d_in[3];
    const int*   bids  = (const int*)  d_in[4];
    const int*   Himg  = (const int*)  d_in[5];
    const int*   Wimg  = (const int*)  d_in[6];
    const float* w1    = (const float*)d_in[7];
    const float* b1    = (const float*)d_in[8];
    const float* w2    = (const float*)d_in[9];
    const float* b2    = (const float*)d_in[10];
    const float* w3    = (const float*)d_in[11];
    const float* b3    = (const float*)d_in[12];

    const int M = in_sizes[4];                    // 4096 matches

    ushort_t* w1b = (ushort_t*)d_ws;              // 128*4608 bf16
    ushort_t* w2b = w1b + (size_t)OC1 * K1;       // 64*1152 bf16 (ws ~1.33 MB)

    const int convThreads = OC1 * K1 + OC2 * K2;
    convert_weights_kernel<<<(convThreads + 255) / 256, 256, 0, stream>>>(w1, w2, w1b, w2b);

    const int nblocks = (M + MPW - 1) / MPW;
    fine_matching_fused<<<nblocks, 256, SMEM_BYTES, stream>>>(
        feat0, feat1, mk0, mk1, bids, Himg, Wimg,
        w1b, b1, w2b, b2, w3, b3, (float*)d_out, M);
}
